// MPNN_3298534884218
// MI455X (gfx1250) — matrix operand order repaired and checked by reading, not compiled, not run on hardware
//
#include <hip/hip_runtime.h>

// ---------------------------------------------------------------------------
// Problem constants (match reference)
// ---------------------------------------------------------------------------
#define NN      20000
#define EENV    400000
#define EEDGE   200000
#define DNODE   128
#define NBASIS  8
#define NLAYER  3
#define RMAX    5.0f
#define LNEPS   1e-5f

// WMMA tiling constants
#define MT 32         // edges per block (two 16-row M-tiles)
#define NT 8          // 128 / 16 N-tiles
#define KS_GATE 12    // 384 / 32
#define KS_E1   9     // ceil(264/32)  (K padded to 288)
#define KS_E2   4     // 128 / 32
#define KP_GATE 384
#define KP_E1   288

typedef __attribute__((ext_vector_type(16))) __bf16 v16bf;
typedef __attribute__((ext_vector_type(8)))  float  v8f;

// ---------------------------------------------------------------------------
// Helpers
// ---------------------------------------------------------------------------
__device__ __forceinline__ unsigned short f2bf(float f) {
    unsigned int u = __float_as_uint(f);
    return (unsigned short)((u + 0x7FFFu + ((u >> 16) & 1u)) >> 16);
}

__device__ __forceinline__ float sigmoid_f(float x) {
    return 1.0f / (1.0f + __expf(-x));
}

__device__ __forceinline__ float softplus_f(float x) {
    // stable: log1p(exp(-|x|)) + max(x,0)
    return log1pf(__expf(-fabsf(x))) + fmaxf(x, 0.0f);
}

union FragU { v16bf v; uint4 q[2]; };

// A fragment: lane (l&15) = row, hi half selects K sub-blocks {0..15} vs {16..31}
__device__ __forceinline__ v16bf load_a(const unsigned short* row, int k0, int hi) {
    FragU u;
    u.q[0] = *(const uint4*)(row + k0 + hi * 16);
    u.q[1] = *(const uint4*)(row + k0 + hi * 16 + 8);
    return u.v;
}

// B fragment from pre-swizzled global weights: contiguous 32B per lane
__device__ __forceinline__ v16bf load_b(const unsigned short* __restrict__ W,
                                        int t, int s, int KS, int lane) {
    FragU u;
    const unsigned short* p = W + ((size_t)((t * KS + s) * 32 + lane) << 4);
    u.q[0] = *(const uint4*)(p);
    u.q[1] = *(const uint4*)(p + 8);
    return u.v;
}

__device__ __forceinline__ v8f wmma_bf16(v16bf a, v16bf b, v8f c) {
    return __builtin_amdgcn_wmma_f32_16x16x32_bf16(
        /*neg_a=*/false, a, /*neg_b=*/false, b,
        /*c_mod=*/(short)0, c, /*reuse_a=*/false, /*reuse_b=*/false);
}

// ---------------------------------------------------------------------------
// Weight swizzle: W[K x 128] (row-major, f32) -> bf16 WMMA B-fragment layout
// dst[((t*KS+s)*32 + l)*16 + j] = W[(s*32 + (l>>4)*16 + j) * 128 + t*16 + (l&15)]
// ---------------------------------------------------------------------------
__global__ __launch_bounds__(256)
void swz_kernel(const float* __restrict__ W, unsigned short* __restrict__ dst,
                int K, int KS) {
    int id = blockIdx.x * 256 + threadIdx.x;
    int total = NT * KS * 32;
    if (id >= total) return;
    int l  = id & 31;
    int ts = id >> 5;
    int s  = ts % KS;
    int t  = ts / KS;
    int n  = t * 16 + (l & 15);
    int kb = s * 32 + ((l >> 4) << 4);
    unsigned short* d = dst + ((size_t)id << 4);
#pragma unroll
    for (int j = 0; j < 16; ++j) {
        int k = kb + j;
        d[j] = (k < K) ? f2bf(W[(size_t)k * DNODE + n]) : (unsigned short)0;
    }
}

// ---------------------------------------------------------------------------
// node[i][d] = W_node[atom_types[i]][d] + b_node[d]
// ---------------------------------------------------------------------------
__global__ __launch_bounds__(256)
void node_init_kernel(const int* __restrict__ at, const float* __restrict__ Wn,
                      const float* __restrict__ bn, float* __restrict__ node) {
    int id = blockIdx.x * 256 + threadIdx.x;
    if (id >= NN * DNODE) return;
    int i = id >> 7, d = id & 127;
    node[id] = Wn[(size_t)at[i] * DNODE + d] + bn[d];
}

// ---------------------------------------------------------------------------
// Per-edge bessel basis + decay
// ---------------------------------------------------------------------------
__global__ __launch_bounds__(256)
void feats_kernel(const float* __restrict__ len, const float* __restrict__ bw,
                  float* __restrict__ bess, float* __restrict__ dec, int E) {
    int e = blockIdx.x * 256 + threadIdx.x;
    if (e >= E) return;
    float x = len[e];
    float inv = 1.0f / x;
#pragma unroll
    for (int k = 0; k < NBASIS; ++k)
        bess[(size_t)e * NBASIS + k] = (2.0f / RMAX) * __sinf(bw[k] * x * (1.0f / RMAX)) * inv;
    dec[e] = __expf(-x * x * (1.0f / 18.0f));
}

// ---------------------------------------------------------------------------
// Edge embedding MLP: tanh([node[i0],node[i1],bessel] @ We1 + b1) @ We2 + b2
// 32-edge tile per block; 8 waves, each owns a 16-wide N-tile x two M-tiles.
// ---------------------------------------------------------------------------
__global__ __launch_bounds__(256)
void embed_kernel(const float* __restrict__ node, const float* __restrict__ bess,
                  const int* __restrict__ i0p, const int* __restrict__ i1p,
                  const unsigned short* __restrict__ W1, const float* __restrict__ b1,
                  const unsigned short* __restrict__ W2, const float* __restrict__ b2,
                  float* __restrict__ feat, int E) {
    __shared__ alignas(16) unsigned short xs[MT][KP_E1];
    __shared__ alignas(16) unsigned short hs[MT][DNODE];
    __shared__ int i0s[MT], i1s[MT];

    const int tid = threadIdx.x;
    const int e0  = blockIdx.x * MT;

    if (tid < MT) {
        int e = e0 + tid;
        i0s[tid] = (e < E) ? i0p[e] : 0;
        i1s[tid] = (e < E) ? i1p[e] : 0;
    }
    __syncthreads();

    for (int j = tid; j < MT * KP_E1; j += 256) {
        int m = j / KP_E1, c = j - m * KP_E1;
        float v = 0.0f;
        int e = e0 + m;
        if (e < E) {
            if (c < 128)      v = node[(size_t)i0s[m] * DNODE + c];
            else if (c < 256) v = node[(size_t)i1s[m] * DNODE + (c - 128)];
            else if (c < 264) v = bess[(size_t)e * NBASIS + (c - 256)];
        }
        xs[m][c] = f2bf(v);
    }
    __syncthreads();

    const int w = tid >> 5, lane = tid & 31;
    const int hi = lane >> 4, lo = lane & 15;
    const unsigned short* xr0 = &xs[lo][0];
    const unsigned short* xr1 = &xs[16 + lo][0];

    v8f acc0 = {}, acc1 = {};
#pragma unroll
    for (int s = 0; s < KS_E1; ++s) {
        v16bf a0 = load_a(xr0, s * 32, hi);
        v16bf a1 = load_a(xr1, s * 32, hi);
        v16bf b  = load_b(W1, w, s, KS_E1, lane);
        acc0 = wmma_bf16(a0, b, acc0);
        acc1 = wmma_bf16(a1, b, acc1);
    }
    const int col = w * 16 + lo;
    const float b1v = b1[col];
#pragma unroll
    for (int i = 0; i < 8; ++i) {
        int m = hi * 8 + i;
        hs[m][col]      = f2bf(tanhf(acc0[i] + b1v));
        hs[16 + m][col] = f2bf(tanhf(acc1[i] + b1v));
    }
    __syncthreads();

    const unsigned short* hr0 = &hs[lo][0];
    const unsigned short* hr1 = &hs[16 + lo][0];
    v8f o0 = {}, o1 = {};
#pragma unroll
    for (int s = 0; s < KS_E2; ++s) {
        v16bf a0 = load_a(hr0, s * 32, hi);
        v16bf a1 = load_a(hr1, s * 32, hi);
        v16bf b  = load_b(W2, w, s, KS_E2, lane);
        o0 = wmma_bf16(a0, b, o0);
        o1 = wmma_bf16(a1, b, o1);
    }
    const float b2v = b2[col];
#pragma unroll
    for (int i = 0; i < 8; ++i) {
        int m = hi * 8 + i;
        int e = e0 + m;
        if (e < E)      feat[(size_t)e * DNODE + col]        = o0[i] + b2v;
        if (e + 16 < E) feat[(size_t)(e + 16) * DNODE + col] = o1[i] + b2v;
    }
}

// ---------------------------------------------------------------------------
// Message kernel: z=[node[i0],node[i1],env]; msg = LN(gate(z)*dec);
// node_out[i1] += msg (atomic scatter-add). node_out must be a copy of node_in.
// ---------------------------------------------------------------------------
__global__ __launch_bounds__(256)
void msg_kernel(const float* __restrict__ node_in, float* __restrict__ node_out,
                const float* __restrict__ env,
                const int* __restrict__ ei0, const int* __restrict__ ei1,
                const float* __restrict__ dec,
                const unsigned short* __restrict__ Wf, const float* __restrict__ bf,
                const unsigned short* __restrict__ Ws, const float* __restrict__ bs,
                const float* __restrict__ lng, const float* __restrict__ lnb, int E) {
    __shared__ alignas(16) unsigned short xs[MT][KP_GATE];
    __shared__ float sg[MT][DNODE];
    __shared__ int i0s[MT], i1s[MT];
    __shared__ float decs[MT], mu[MT], rsg[MT];

    const int tid = threadIdx.x;
    const int e0  = blockIdx.x * MT;

    if (tid < MT) {
        int e = e0 + tid;
        if (e < E) { i0s[tid] = ei0[e]; i1s[tid] = ei1[e]; decs[tid] = dec[e]; }
        else       { i0s[tid] = 0;      i1s[tid] = 0;      decs[tid] = 0.0f;  }
    }
    __syncthreads();

    for (int j = tid; j < MT * KP_GATE; j += 256) {
        int m = j / KP_GATE, c = j - m * KP_GATE;
        float v = 0.0f;
        int e = e0 + m;
        if (e < E) {
            if (c < 128)      v = node_in[(size_t)i0s[m] * DNODE + c];
            else if (c < 256) v = node_in[(size_t)i1s[m] * DNODE + (c - 128)];
            else              v = env[(size_t)e * DNODE + (c - 256)];
        }
        xs[m][c] = f2bf(v);
    }
    __syncthreads();

    const int w = tid >> 5, lane = tid & 31;
    const int hi = lane >> 4, lo = lane & 15;
    const unsigned short* xr0 = &xs[lo][0];
    const unsigned short* xr1 = &xs[16 + lo][0];

    v8f f0 = {}, f1 = {}, s0 = {}, s1 = {};
#pragma unroll
    for (int s = 0; s < KS_GATE; ++s) {
        v16bf a0 = load_a(xr0, s * 32, hi);
        v16bf a1 = load_a(xr1, s * 32, hi);
        v16bf bF = load_b(Wf, w, s, KS_GATE, lane);
        v16bf bS = load_b(Ws, w, s, KS_GATE, lane);
        f0 = wmma_bf16(a0, bF, f0);
        f1 = wmma_bf16(a1, bF, f1);
        s0 = wmma_bf16(a0, bS, s0);
        s1 = wmma_bf16(a1, bS, s1);
    }
    const int col = w * 16 + lo;
    const float bfv = bf[col], bsv = bs[col];
#pragma unroll
    for (int i = 0; i < 8; ++i) {
        int m = hi * 8 + i;
        sg[m][col]      = sigmoid_f(f0[i] + bfv) * softplus_f(s0[i] + bsv) * decs[m];
        sg[16 + m][col] = sigmoid_f(f1[i] + bfv) * softplus_f(s1[i] + bsv) * decs[16 + m];
    }
    __syncthreads();

    if (tid < MT) {
        float su = 0.0f, sq = 0.0f;
#pragma unroll 8
        for (int c = 0; c < DNODE; ++c) { float v = sg[tid][c]; su += v; sq += v * v; }
        float m_ = su * (1.0f / DNODE);
        float var = sq * (1.0f / DNODE) - m_ * m_;
        mu[tid]  = m_;
        rsg[tid] = rsqrtf(var + LNEPS);
    }
    __syncthreads();

    for (int j = tid; j < MT * DNODE; j += 256) {
        int m = j >> 7, c = j & 127;
        if (e0 + m < E) {
            float v = (sg[m][c] - mu[m]) * rsg[m] * lng[c] + lnb[c];
            atomicAdd(&node_out[(size_t)i1s[m] * DNODE + c], v);
        }
    }
}

// ---------------------------------------------------------------------------
// Gated update: z=[node[i0],feat,node[i1]]; feat_out = gate(z)*dec (in-place ok)
// ---------------------------------------------------------------------------
__global__ __launch_bounds__(256)
void gate_kernel(const float* __restrict__ node,
                 const float* __restrict__ feat_in, float* __restrict__ feat_out,
                 const int* __restrict__ i0p, const int* __restrict__ i1p,
                 const float* __restrict__ dec,
                 const unsigned short* __restrict__ Wf, const float* __restrict__ bf,
                 const unsigned short* __restrict__ Ws, const float* __restrict__ bs,
                 int E) {
    __shared__ alignas(16) unsigned short xs[MT][KP_GATE];
    __shared__ int i0s[MT], i1s[MT];
    __shared__ float decs[MT];

    const int tid = threadIdx.x;
    const int e0  = blockIdx.x * MT;

    if (tid < MT) {
        int e = e0 + tid;
        if (e < E) { i0s[tid] = i0p[e]; i1s[tid] = i1p[e]; decs[tid] = dec[e]; }
        else       { i0s[tid] = 0;      i1s[tid] = 0;      decs[tid] = 0.0f;  }
    }
    __syncthreads();

    for (int j = tid; j < MT * KP_GATE; j += 256) {
        int m = j / KP_GATE, c = j - m * KP_GATE;
        float v = 0.0f;
        int e = e0 + m;
        if (e < E) {
            if (c < 128)      v = node[(size_t)i0s[m] * DNODE + c];
            else if (c < 256) v = feat_in[(size_t)e * DNODE + (c - 128)];
            else              v = node[(size_t)i1s[m] * DNODE + (c - 256)];
        }
        xs[m][c] = f2bf(v);
    }
    __syncthreads();

    const int w = tid >> 5, lane = tid & 31;
    const int hi = lane >> 4, lo = lane & 15;
    const unsigned short* xr0 = &xs[lo][0];
    const unsigned short* xr1 = &xs[16 + lo][0];

    v8f f0 = {}, f1 = {}, s0 = {}, s1 = {};
#pragma unroll
    for (int s = 0; s < KS_GATE; ++s) {
        v16bf a0 = load_a(xr0, s * 32, hi);
        v16bf a1 = load_a(xr1, s * 32, hi);
        v16bf bF = load_b(Wf, w, s, KS_GATE, lane);
        v16bf bS = load_b(Ws, w, s, KS_GATE, lane);
        f0 = wmma_bf16(a0, bF, f0);
        f1 = wmma_bf16(a1, bF, f1);
        s0 = wmma_bf16(a0, bS, s0);
        s1 = wmma_bf16(a1, bS, s1);
    }
    const int col = w * 16 + lo;
    const float bfv = bf[col], bsv = bs[col];
#pragma unroll
    for (int i = 0; i < 8; ++i) {
        int m = hi * 8 + i;
        int e = e0 + m;
        float g0 = sigmoid_f(f0[i] + bfv) * softplus_f(s0[i] + bsv) * decs[m];
        float g1 = sigmoid_f(f1[i] + bfv) * softplus_f(s1[i] + bsv) * decs[16 + m];
        if (e < E)      feat_out[(size_t)e * DNODE + col]        = g0;
        if (e + 16 < E) feat_out[(size_t)(e + 16) * DNODE + col] = g1;
    }
}

// ---------------------------------------------------------------------------
// Host launcher
// ---------------------------------------------------------------------------
extern "C" void kernel_launch(void* const* d_in, const int* in_sizes, int n_in,
                              void* d_out, int out_size, void* d_ws, size_t ws_size,
                              hipStream_t stream) {
    (void)in_sizes; (void)n_in; (void)out_size; (void)ws_size;

    const int*   at    = (const int*)  d_in[0];
    const int*   envi  = (const int*)  d_in[1];
    const int*   edgi  = (const int*)  d_in[2];
    const float* envl  = (const float*)d_in[3];
    const float* edgl  = (const float*)d_in[4];
    const float* Wnode = (const float*)d_in[5];
    const float* bnode = (const float*)d_in[6];
    const float* We1   = (const float*)d_in[7];
    const float* be1   = (const float*)d_in[8];
    const float* We2   = (const float*)d_in[9];
    const float* be2   = (const float*)d_in[10];
    const float* bw    = (const float*)d_in[11];
    const float* Wnf   = (const float*)d_in[12];
    const float* bnf   = (const float*)d_in[13];
    const float* Wns   = (const float*)d_in[14];
    const float* bns   = (const float*)d_in[15];
    const float* Wef   = (const float*)d_in[16];
    const float* bef   = (const float*)d_in[17];
    const float* Wes   = (const float*)d_in[18];
    const float* bes   = (const float*)d_in[19];
    const float* lng   = (const float*)d_in[20];
    const float* lnb   = (const float*)d_in[21];

    // --- workspace carve-up ---
    char* ws = (char*)d_ws;
    size_t off = 0;
    auto take = [&](size_t bytes) -> char* {
        char* p = ws + off;
        off = (off + bytes + 255) & ~(size_t)255;
        return p;
    };
    const size_t SW_E1 = (size_t)NT * KS_E1   * 512;  // ushort elems
    const size_t SW_E2 = (size_t)NT * KS_E2   * 512;
    const size_t SW_G  = (size_t)NT * KS_GATE * 512;  // per matrix per layer

    unsigned short* we1s = (unsigned short*)take(SW_E1 * 2);
    unsigned short* we2s = (unsigned short*)take(SW_E2 * 2);
    unsigned short* wnfs = (unsigned short*)take(NLAYER * SW_G * 2);
    unsigned short* wnss = (unsigned short*)take(NLAYER * SW_G * 2);
    unsigned short* wefs = (unsigned short*)take(NLAYER * SW_G * 2);
    unsigned short* wess = (unsigned short*)take(NLAYER * SW_G * 2);
    float* bessE = (float*)take((size_t)EENV  * NBASIS * 4);
    float* bessB = (float*)take((size_t)EEDGE * NBASIS * 4);
    float* decE  = (float*)take((size_t)EENV  * 4);
    float* decB  = (float*)take((size_t)EEDGE * 4);
    float* envf  = (float*)take((size_t)EENV  * DNODE * 4);
    float* nodeA = (float*)take((size_t)NN    * DNODE * 4);

    float* out_node = (float*)d_out;
    float* out_edge = out_node + (size_t)NN * DNODE;

    // --- weight prep (bf16 + WMMA fragment swizzle) ---
    swz_kernel<<<(NT * KS_E1 * 32 + 255) / 256, 256, 0, stream>>>(We1, we1s, 264, KS_E1);
    swz_kernel<<<(NT * KS_E2 * 32 + 255) / 256, 256, 0, stream>>>(We2, we2s, 128, KS_E2);
    const int gblk = (NT * KS_GATE * 32 + 255) / 256;
    for (int l = 0; l < NLAYER; ++l) {
        size_t wsrc = (size_t)l * KP_GATE * DNODE;
        size_t wdst = (size_t)l * SW_G;
        swz_kernel<<<gblk, 256, 0, stream>>>(Wnf + wsrc, wnfs + wdst, KP_GATE, KS_GATE);
        swz_kernel<<<gblk, 256, 0, stream>>>(Wns + wsrc, wnss + wdst, KP_GATE, KS_GATE);
        swz_kernel<<<gblk, 256, 0, stream>>>(Wef + wsrc, wefs + wdst, KP_GATE, KS_GATE);
        swz_kernel<<<gblk, 256, 0, stream>>>(Wes + wsrc, wess + wdst, KP_GATE, KS_GATE);
    }

    // --- node init + per-edge features ---
    node_init_kernel<<<(NN * DNODE + 255) / 256, 256, 0, stream>>>(at, Wnode, bnode, nodeA);
    feats_kernel<<<(EENV  + 255) / 256, 256, 0, stream>>>(envl, bw, bessE, decE, EENV);
    feats_kernel<<<(EEDGE + 255) / 256, 256, 0, stream>>>(edgl, bw, bessB, decB, EEDGE);

    // --- edge embeddings ---
    const int tE = (EENV + MT - 1) / MT, tB = (EEDGE + MT - 1) / MT;
    embed_kernel<<<tE, 256, 0, stream>>>(nodeA, bessE, envi, envi + EENV,
                                         we1s, be1, we2s, be2, envf, EENV);
    embed_kernel<<<tB, 256, 0, stream>>>(nodeA, bessB, edgi, edgi + EEDGE,
                                         we1s, be1, we2s, be2, out_edge, EEDGE);

    // --- message-passing layers (node ping-pong: final lands in d_out) ---
    float* cur = nodeA;
    float* nxt = out_node;
    for (int l = 0; l < NLAYER; ++l) {
        size_t wdst = (size_t)l * SW_G;
        hipMemcpyAsync(nxt, cur, (size_t)NN * DNODE * 4, hipMemcpyDeviceToDevice, stream);
        msg_kernel<<<tE, 256, 0, stream>>>(cur, nxt, envf, envi, envi + EENV, decE,
                                           wnfs + wdst, bnf + l * DNODE,
                                           wnss + wdst, bns + l * DNODE,
                                           lng + l * DNODE, lnb + l * DNODE, EENV);
        gate_kernel<<<tE, 256, 0, stream>>>(nxt, envf, envf, envi, envi + EENV, decE,
                                            wefs + wdst, bef + l * DNODE,
                                            wess + wdst, bes + l * DNODE, EENV);
        gate_kernel<<<tB, 256, 0, stream>>>(nxt, out_edge, out_edge, edgi, edgi + EEDGE, decB,
                                            wefs + wdst, bef + l * DNODE,
                                            wess + wdst, bes + l * DNODE, EEDGE);
        float* t = cur; cur = nxt; nxt = t;
    }
}
